// ModelRQuery_5806795784426
// MI455X (gfx1250) — matrix operand order repaired and checked by reading, not compiled, not run on hardware
//
#include <hip/hip_runtime.h>
#include <hip/hip_bf16.h>
#include <math.h>

typedef __bf16 bfx;
typedef __attribute__((ext_vector_type(16))) __bf16 v16bf;
typedef __attribute__((ext_vector_type(8)))  __bf16 v8bf;
typedef __attribute__((ext_vector_type(8)))  float  v8f;
typedef __attribute__((ext_vector_type(4))) unsigned int u32x4;
typedef __attribute__((ext_vector_type(8)))  int i32x8;
typedef __attribute__((ext_vector_type(4)))  int i32x4;

#define D      1024
#define TWO_D  2048
#define C_OUT  53
#define NPB    64
#define NBAGS  512
#define STEPS  63

#if defined(__HIP_DEVICE_COMPILE__) && __has_builtin(__builtin_amdgcn_tensor_load_to_lds) && __has_builtin(__builtin_amdgcn_s_wait_tensorcnt)
#define HAVE_TDM 1
#else
#define HAVE_TDM 0
#endif

__device__ inline v16bf combine8(v8bf lo, v8bf hi) {
  v16bf r;
#pragma unroll
  for (int i = 0; i < 8; ++i) { r[i] = lo[i]; r[8 + i] = hi[i]; }
  return r;
}

__device__ inline v16bf frag_from_f32(const float* p0, const float* p1) {
  v16bf r;
#pragma unroll
  for (int i = 0; i < 8; ++i) { r[i] = (__bf16)p0[i]; r[8 + i] = (__bf16)p1[i]; }
  return r;
}

// ---------------------------------------------------------------------------
// TDM: issue TENSOR_LOAD_TO_LDS of a 2D bf16 tile (tile0 x tile1 elements,
// row stride tdim0) from gsrc into LDS offset lds_off. D# packed per
// CDNA5 ISA ch.8: group0 = {count=1, lds_addr, global_addr, type=2},
// group1 = {data_size=2B, tensor_dim0/1, tile_dim0/1, stride0}.
// 6-arg toolchain variant: (g0, g1, g2, g3, g4, cpol) with groups 2..4 zero
// for 2D tensors.
// ---------------------------------------------------------------------------
__device__ inline void tdm_load_tile(const bfx* gsrc, unsigned lds_off,
                                     int tdim0, int tdim1, int tile0, int tile1) {
#if HAVE_TDM
  unsigned long long ga = (unsigned long long)(const void*)gsrc;
  u32x4 g0;
  g0[0] = 1u;                                            // count=1 (valid), user mode
  g0[1] = lds_off;                                       // lds_addr (bytes)
  g0[2] = (unsigned)ga;                                  // global_addr[31:0]
  g0[3] = (unsigned)((ga >> 32) & 0x1FFFFFFull) | 0x80000000u;  // addr[56:32] | type=2
  i32x8 g1;
  g1[0] = (int)(1u << 16);                               // data_size = 1 -> 2 bytes
  g1[1] = (int)(((unsigned)tdim0 & 0xFFFFu) << 16);      // tensor_dim0[15:0]
  g1[2] = (int)((((unsigned)tdim0 >> 16) & 0xFFFFu) |
                (((unsigned)tdim1 & 0xFFFFu) << 16));    // tensor_dim0[31:16], dim1[15:0]
  g1[3] = (int)((((unsigned)tdim1 >> 16) & 0xFFFFu) |
                ((unsigned)tile0 << 16));                // tensor_dim1[31:16], tile_dim0
  g1[4] = (int)(unsigned)tile1;                          // tile_dim1 (tile_dim2 = 0)
  g1[5] = (int)(unsigned)tdim0;                          // tensor_dim0_stride[31:0]
  g1[6] = 0;                                             // stride0[47:32], stride1[15:0]
  g1[7] = 0;                                             // stride1[47:16]
  i32x4 z4 = {0, 0, 0, 0};
  i32x8 z8 = {0, 0, 0, 0, 0, 0, 0, 0};
  __builtin_amdgcn_tensor_load_to_lds(g0, g1, z4, z4, z8, 0);
#else
  (void)gsrc; (void)lds_off; (void)tdim0; (void)tdim1; (void)tile0; (void)tile1;
#endif
}

// ---------------------------------------------------------------------------
// Convert fc*_w (f32, [N=1024][K]) into bf16 swizzled WMMA-B tile layout.
// ---------------------------------------------------------------------------
__global__ void prep_weights(const float* __restrict__ src, bfx* __restrict__ dst,
                             int K, int KT) {
  int g = blockIdx.x * blockDim.x + threadIdx.x;   // total = K * 1024
  int tile   = g >> 9;
  int within = g & 511;
  int lane = within >> 4;
  int e    = within & 15;
  int nt = tile / KT;
  int kt = tile - nt * KT;
  int n = nt * 16 + (lane & 15);
  int k = kt * 32 + ((e >> 3) << 4) + ((lane >> 4) << 3) + (e & 7);
  dst[g] = (bfx)src[(size_t)n * K + k];
}

// ---------------------------------------------------------------------------
// Per-bag: 64x64 Gram via WMMA bf16 (K=1024), cosine-sum distance, softmax
// into Huffman weights; init alive; copy rep -> feats (bf16).
// ---------------------------------------------------------------------------
__global__ void __launch_bounds__(256)
gram_init(const float* __restrict__ rep, float* __restrict__ wts,
          int* __restrict__ alive, bfx* __restrict__ feats) {
  const int bag  = blockIdx.x;
  const int tid  = threadIdx.x;
  const int wave = tid >> 5, lane = tid & 31;
  const float* S = rep + (size_t)bag * NPB * D;

  __shared__ float gram[NPB][NPB + 1];
  __shared__ float nrm[NPB];
  __shared__ float dist[NPB];

  const int tm  = wave >> 1;
  const int tn0 = (wave & 1) * 2;

  v8f acc0 = {}; v8f acc1 = {};
  const int arow  = tm * 16 + (lane & 15);
  const int kb    = (lane >> 4) * 16;
  const int brow0 = tn0 * 16 + (lane & 15);
  const int kbf   = (lane >> 4) * 16;

  for (int kt = 0; kt < D / 32; ++kt) {
    const int k0 = kt * 32;
    v16bf a  = frag_from_f32(S + (size_t)arow * D + k0 + kb,
                             S + (size_t)arow * D + k0 + kb + 8);
    v16bf b0 = frag_from_f32(S + (size_t)brow0 * D + k0 + kbf,
                             S + (size_t)brow0 * D + k0 + kbf + 8);
    v16bf b1 = frag_from_f32(S + (size_t)(brow0 + 16) * D + k0 + kbf,
                             S + (size_t)(brow0 + 16) * D + k0 + kbf + 8);
    acc0 = __builtin_amdgcn_wmma_f32_16x16x32_bf16(false, a, false, b0,
                                                   (short)0, acc0, false, false);
    acc1 = __builtin_amdgcn_wmma_f32_16x16x32_bf16(false, a, false, b1,
                                                   (short)0, acc1, false, false);
  }

#pragma unroll
  for (int v = 0; v < 8; ++v) {
    const int r = tm * 16 + v + (lane >> 4) * 8;
    gram[r][tn0 * 16 + (lane & 15)]       = acc0[v];
    gram[r][(tn0 + 1) * 16 + (lane & 15)] = acc1[v];
  }
  __syncthreads();

  if (tid < NPB) nrm[tid] = sqrtf(fmaxf(gram[tid][tid], 0.f));
  __syncthreads();
  if (tid < NPB) {
    float s = 0.f;
    for (int r = 0; r < NPB; ++r)
      s += gram[r][tid] / fmaxf(nrm[r] * nrm[tid], 1e-8f);
    dist[tid] = s;
  }
  __syncthreads();
  if (tid < NPB) {
    float m = -1e30f;
    for (int j = 0; j < NPB; ++j) m = fmaxf(m, dist[j]);
    float denom = 0.f;
    for (int j = 0; j < NPB; ++j) denom += expf(dist[j] - m);
    wts[bag * NPB + tid]   = expf(dist[tid] - m) / denom;
    alive[bag * NPB + tid] = 1;
  }

  bfx* fb = feats + (size_t)bag * NPB * D;
  for (int i = tid; i < NPB * D; i += 256) fb[i] = (bfx)S[i];
}

// ---------------------------------------------------------------------------
// Per step: pick two smallest alive weights per bag, update weights/alive,
// record indices, gather X[bag] = tanh(concat(feats[i1], feats[i2])).
// ---------------------------------------------------------------------------
__global__ void select_gather(float* __restrict__ wts, int* __restrict__ alive,
                              int* __restrict__ idx, const bfx* __restrict__ feats,
                              bfx* __restrict__ X) {
  const int bag = blockIdx.x;
  const int tid = threadIdx.x;
  __shared__ int si1, si2;
  if (tid == 0) {
    const float* w  = wts + bag * NPB;
    const int*   al = alive + bag * NPB;
    float b1 = 3.4e38f, b2 = 3.4e38f;
    int i1 = 0, i2 = 1;
    for (int j = 0; j < NPB; ++j) {
      if (!al[j]) continue;
      const float v = w[j];
      if (v < b1)      { b2 = b1; i2 = i1; b1 = v; i1 = j; }
      else if (v < b2) { b2 = v; i2 = j; }
    }
    idx[bag * 2]     = i1;
    idx[bag * 2 + 1] = i2;
    wts[bag * NPB + i1] = b1 + b2;
    alive[bag * NPB + i2] = 0;
    si1 = i1; si2 = i2;
  }
  __syncthreads();
  const bfx* f1 = feats + ((size_t)bag * NPB + si1) * D;
  const bfx* f2 = feats + ((size_t)bag * NPB + si2) * D;
  bfx* xr = X + (size_t)bag * TWO_D;
  for (int i = tid; i < D; i += 64) {
    xr[i]     = (bfx)tanhf((float)f1[i]);
    xr[D + i] = (bfx)tanhf((float)f2[i]);
  }
}

// ---------------------------------------------------------------------------
// Batched GEMM: Out = epilogue(A(512xK bf16) @ Ws(swizzled KxD bf16) + bias).
// Block tile 64(M) x 128(N), 8 waves. A-panel staged into LDS by the TDM
// (64x128 bf16 chunks, double-buffered, TENSORcnt + barrier sync); B frags
// stream from L2 out of the pre-swizzled weight layout.
// MODE 0: Y = bf16(tanh(acc + fc1_b))  (FC1)
// MODE 1: feats[bag][idx[bag].i1] = bf16(acc + fc2_b)  (FC2 + scatter-back)
// ---------------------------------------------------------------------------
template <int K, int MODE>
__global__ void __launch_bounds__(256)
gemm_bf16(const bfx* __restrict__ A, const bfx* __restrict__ Ws,
          const float* __restrict__ bias, bfx* __restrict__ Yout,
          bfx* __restrict__ feats, const int* __restrict__ idx) {
  constexpr int KT = K / 32;    // 16x16x32 steps
  constexpr int NC = K / 128;   // TDM chunks (4 wmma K-steps each)
  const int tid  = threadIdx.x;
  const int wave = tid >> 5, lane = tid & 31;
  const int mbase = blockIdx.y * 64 + (wave & 3) * 16;
  const int nbase = blockIdx.x * 128 + (wave >> 2) * 64;

  __shared__ __align__(16) bfx atile[2][64 * 128];
  const unsigned lds0 = (unsigned)(size_t)&atile[0][0];
  const unsigned lds1 = (unsigned)(size_t)&atile[1][0];
  const bfx* apanel = A + (size_t)(blockIdx.y * 64) * K;

  const int  nt0   = nbase >> 4;
  const bfx* wbase = Ws + (size_t)nt0 * KT * 512 + lane * 16;

  const int lrow  = (wave & 3) * 16 + (lane & 15);   // row within 64-row panel
  const int kbase = (lane >> 4) * 8;                 // A frag K sub-base

#if HAVE_TDM
  if (wave == 0)
    tdm_load_tile(apanel, lds0, K, NBAGS, 128, 64);
#endif

  v8f acc[4] = {};

  for (int kc = 0; kc < NC; ++kc) {
#if HAVE_TDM
    if (wave == 0) {
      if (kc + 1 < NC) {
        tdm_load_tile(apanel + (kc + 1) * 128, (kc & 1) ? lds0 : lds1,
                      K, NBAGS, 128, 64);
        __builtin_amdgcn_s_wait_tensorcnt(1);  // chunk kc complete
      } else {
        __builtin_amdgcn_s_wait_tensorcnt(0);
      }
    }
#else
    {  // cooperative fallback copy of chunk kc
      bfx* dstb = atile[kc & 1];
      for (int i = tid; i < 64 * 128; i += 256) {
        const int r = i >> 7, c = i & 127;
        dstb[i] = apanel[(size_t)r * K + kc * 128 + c];
      }
    }
#endif
    __syncthreads();

    const bfx* abuf = atile[kc & 1];
#pragma unroll
    for (int kk = 0; kk < 4; ++kk) {
      const int kt = kc * 4 + kk;
      const bfx* ap = abuf + lrow * 128 + kk * 32 + kbase;
      const v8bf alo = *(const v8bf*)ap;         // ds_load_b128
      const v8bf ahi = *(const v8bf*)(ap + 16);  // ds_load_b128
      const v16bf a  = combine8(alo, ahi);
      __builtin_prefetch(wbase + ((size_t)0 * KT + kt + 8) * 512, 0, 0);
#pragma unroll
      for (int t = 0; t < 4; ++t) {
        const bfx* wp = wbase + ((size_t)t * KT + kt) * 512;
        const v8bf blo = *(const v8bf*)wp;
        const v8bf bhi = *(const v8bf*)(wp + 8);
        const v16bf b  = combine8(blo, bhi);
        acc[t] = __builtin_amdgcn_wmma_f32_16x16x32_bf16(false, a, false, b,
                                                         (short)0, acc[t],
                                                         false, false);
      }
    }
    __syncthreads();
  }

#pragma unroll
  for (int t = 0; t < 4; ++t) {
    const int col = nbase + t * 16 + (lane & 15);
    const float bb = bias[col];
#pragma unroll
    for (int v = 0; v < 8; ++v) {
      const int row = mbase + v + (lane >> 4) * 8;
      const float val = acc[t][v] + bb;
      if (MODE == 0) {
        Yout[(size_t)row * D + col] = (bfx)tanhf(val);
      } else {
        const int i1 = idx[row * 2];
        feats[((size_t)row * NPB + i1) * D + col] = (bfx)val;
      }
    }
  }
}

// ---------------------------------------------------------------------------
// Final: root = first alive node; out[bag][c] = sigmoid(feats_root . rel[c]).
// ---------------------------------------------------------------------------
__global__ void final_scores(const bfx* __restrict__ feats,
                             const int* __restrict__ alive,
                             const float* __restrict__ rel,
                             float* __restrict__ out) {
  const int bag = blockIdx.x;
  const int tid = threadIdx.x;
  __shared__ int sroot;
  if (tid == 0) {
    int r = 0;
    for (int j = 0; j < NPB; ++j)
      if (alive[bag * NPB + j]) { r = j; break; }
    sroot = r;
  }
  __syncthreads();
  if (tid < C_OUT) {
    const bfx*  f = feats + ((size_t)bag * NPB + sroot) * D;
    const float* e = rel + (size_t)tid * D;
    float s = 0.f;
    for (int d = 0; d < D; ++d) s += (float)f[d] * e[d];
    out[bag * C_OUT + tid] = 1.f / (1.f + expf(-s));
  }
}

// ---------------------------------------------------------------------------
extern "C" void kernel_launch(void* const* d_in, const int* in_sizes, int n_in,
                              void* d_out, int out_size, void* d_ws, size_t ws_size,
                              hipStream_t stream) {
  const float* rep  = (const float*)d_in[0];
  const float* fc1w = (const float*)d_in[1];
  const float* fc1b = (const float*)d_in[2];
  const float* fc2w = (const float*)d_in[3];
  const float* fc2b = (const float*)d_in[4];
  const float* rel  = (const float*)d_in[5];
  float* out = (float*)d_out;

  // Workspace carve-up (~73.5 MB total)
  bfx* feats = (bfx*)d_ws;                                   // 512*64*1024 bf16
  bfx* X     = feats + (size_t)NBAGS * NPB * D;              // 512*2048
  bfx* Y     = X + (size_t)NBAGS * TWO_D;                    // 512*1024
  bfx* W1s   = Y + (size_t)NBAGS * D;                        // 2048*1024
  bfx* W2s   = W1s + (size_t)TWO_D * D;                      // 1024*1024
  float* wts = (float*)(W2s + (size_t)D * D);                // 512*64
  int* alive = (int*)(wts + NBAGS * NPB);                    // 512*64
  int* idx   = alive + NBAGS * NPB;                          // 512*2

  prep_weights<<<(TWO_D * D) / 256, 256, 0, stream>>>(fc1w, W1s, TWO_D, TWO_D / 32);
  prep_weights<<<(D * D) / 256, 256, 0, stream>>>(fc2w, W2s, D, D / 32);
  gram_init<<<NBAGS, 256, 0, stream>>>(rep, wts, alive, feats);

  const dim3 gg(D / 128, NBAGS / 64);   // 8 x 8 blocks
  for (int t = 0; t < STEPS; ++t) {
    select_gather<<<NBAGS, 64, 0, stream>>>(wts, alive, idx, feats, X);
    gemm_bf16<TWO_D, 0><<<gg, 256, 0, stream>>>(X, W1s, fc1b, Y, nullptr, nullptr);
    gemm_bf16<D, 1><<<gg, 256, 0, stream>>>(Y, W2s, fc2b, nullptr, feats, idx);
  }
  final_scores<<<NBAGS, 64, 0, stream>>>(feats, alive, rel, out);
}
